// GCN_12524124635296
// MI455X (gfx1250) — compile-verified, matrix operand order checked by reading, not run on hardware
//
#include <hip/hip_runtime.h>
#include <hip/hip_bf16.h>
#include <math.h>

// ---------------------------------------------------------------------------
// GCN + TopK pooling pipeline for MI455X (gfx1250, wave32, WMMA).
//
// Sizes (compile-time constants from the reference):
//   N0 = 65536 nodes, E = 1,048,576 edges, dims 128 -> 128 -> 64 -> 32 -> 16
// ---------------------------------------------------------------------------

#define N0_NODES 65536
#define E_EDGES  1048576
#define B_GRAPHS 64

typedef __attribute__((ext_vector_type(16))) __bf16 v16bf;
typedef __attribute__((ext_vector_type(8)))  float  v8f;

union BFrag {
    v16bf v;
    uint4 q[2];
};

// ---------------------------------------------------------------------------
// WMMA GEMM:  C[M x Nc] = A[M x K] (bf16, row major) * W[K x Nc]
// W is supplied transposed (WT = Nc x K, bf16, row major) and staged into LDS
// via GLOBAL_LOAD_ASYNC_TO_LDS_B128 (ASYNCcnt-tracked, no VGPR round trip),
// with a +8 element pad per row to rotate LDS banks (stride = (K+8)*2 bytes
// -> 4-bank skew per row -> conflict-free b128 reads).
//
// Block = 256 threads = 8 waves; each wave owns a 16-row strip and all NT
// 16-wide column tiles.  Per k-step we preload ALL NT B fragments into a
// register array so the ds_load_b128 clause drains with partial DScnt waits
// while the WMMA chain (XDL / TRANS class) co-executes.
// ---------------------------------------------------------------------------
template <int K, int NT>
__global__ __launch_bounds__(256)
void gemm_wmma(const __bf16* __restrict__ A, const __bf16* __restrict__ WT,
               float* __restrict__ C) {
    constexpr int Nc   = NT * 16;
    constexpr int LDSK = K + 8;                 // pad: skew LDS banks
    __shared__ __align__(16) __bf16 lds[NT * 16 * LDSK];

    const int tid = threadIdx.x;

    // --- async stage of WT (Nc x K bf16) into padded LDS ------------------
    // 16-byte chunks (8 bf16); K % 8 == 0 so a chunk never crosses a row.
    {
        constexpr int CHUNKS = (Nc * K) / 8;    // 16B chunks
        for (int ch = tid; ch < CHUNKS; ch += 256) {
            int elem = ch * 8;
            int r = elem / K, c = elem - r * K;
            unsigned int ldsoff =
                (unsigned int)(size_t)(&lds[r * LDSK + c]);
            unsigned long long gptr =
                (unsigned long long)(size_t)(WT + elem);
            asm volatile("global_load_async_to_lds_b128 %0, %1, off"
                         :: "v"(ldsoff), "v"(gptr) : "memory");
        }
        asm volatile("s_wait_asynccnt 0x0" ::: "memory");
    }
    __syncthreads();

    const int wave = tid >> 5;
    const int lane = tid & 31;
    const int l16  = lane & 15;
    const int hi   = (lane >> 4) & 1;
    const int row0 = blockIdx.x * 128 + wave * 16;

    v8f acc[NT];
#pragma unroll
    for (int t = 0; t < NT; ++t) acc[t] = (v8f){0.f,0.f,0.f,0.f,0.f,0.f,0.f,0.f};

    // A fragment, contiguous k order (B uses the same): lane l<16 holds row l, K=k0..k0+15; lane
    // l+16 holds row l, K=k0+16..k0+31 (16 contiguous bf16 = two b128 loads).
    const __bf16* arow = A + (size_t)(row0 + l16) * K + hi * 16;
    __builtin_prefetch(arow, 0, 0);            // global_prefetch_b8

#pragma unroll
    for (int k0 = 0; k0 < K; k0 += 32) {
        BFrag af;
        af.q[0] = *(const uint4*)(arow + k0);
        af.q[1] = *(const uint4*)(arow + k0 + 8);

        BFrag bf[NT];
#pragma unroll
        for (int t = 0; t < NT; ++t) {
            const __bf16* bp = &lds[(t * 16 + l16) * LDSK + k0 + hi * 16];
            bf[t].q[0] = *(const uint4*)(bp);
            bf[t].q[1] = *(const uint4*)(bp + 8);
        }
#pragma unroll
        for (int t = 0; t < NT; ++t)
            acc[t] = __builtin_amdgcn_wmma_f32_16x16x32_bf16(
                false, af.v, false, bf[t].v, (short)0, acc[t], false, false);
    }

    // C layout per ISA: VGPR j -> row j (lanes 0-15) / row j+8 (lanes 16-31),
    // column = tile*16 + lane%16.  Stores coalesce across lanes.
#pragma unroll
    for (int t = 0; t < NT; ++t) {
        float* crow = C + (size_t)(row0 + hi * 8) * Nc + t * 16 + l16;
#pragma unroll
        for (int j = 0; j < 8; ++j)
            crow[(size_t)j * Nc] = acc[t][j];
    }
}

// ---------------------------------------------------------------------------
// Elementwise / graph kernels
// ---------------------------------------------------------------------------
__global__ void cvt_bf16_kernel(const float* __restrict__ s, __bf16* __restrict__ d, int n) {
    int i = blockIdx.x * blockDim.x + threadIdx.x;
    if (i < n) d[i] = (__bf16)s[i];
}

// W (K x N, f32) -> WT (N x K, bf16)
__global__ void cvt_wt_kernel(const float* __restrict__ W, __bf16* __restrict__ WT,
                              int K, int N) {
    int t = blockIdx.x * blockDim.x + threadIdx.x;
    if (t >= K * N) return;
    int k = t / N, c = t - k * N;
    WT[(size_t)c * K + k] = (__bf16)W[t];
}

__global__ void fill_f32_kernel(float* p, float v, int n) {
    int i = blockIdx.x * blockDim.x + threadIdx.x;
    if (i < n) p[i] = v;
}
__global__ void fill_i32_kernel(int* p, int v, int n) {
    int i = blockIdx.x * blockDim.x + threadIdx.x;
    if (i < n) p[i] = v;
}

__global__ void init_edges_kernel(const int* __restrict__ ei, int* __restrict__ src,
                                  int* __restrict__ dst, int* __restrict__ em, int E) {
    int e = blockIdx.x * blockDim.x + threadIdx.x;
    if (e >= E) return;
    src[e] = ei[e];
    dst[e] = ei[E + e];
    em[e]  = 1;
}

__global__ void deg_accum_kernel(const int* __restrict__ dst, const int* __restrict__ em,
                                 float* __restrict__ deg, int E) {
    int e = blockIdx.x * blockDim.x + threadIdx.x;
    if (e >= E) return;
    if (em[e]) atomicAdd(&deg[dst[e]], 1.0f);
}

__global__ void rsqrt_kernel(float* __restrict__ dis, const float* __restrict__ deg, int n) {
    int i = blockIdx.x * blockDim.x + threadIdx.x;
    if (i < n) dis[i] = rsqrtf(deg[i]);
}

// One thread per (edge, feature); coalesced gather + float atomic scatter.
__global__ void agg_kernel(const float* __restrict__ h, const int* __restrict__ src,
                           const int* __restrict__ dst, const int* __restrict__ em,
                           const float* __restrict__ dis, float* __restrict__ agg,
                           int E, int Fsh) {
    long long idx = (long long)blockIdx.x * blockDim.x + threadIdx.x;
    int e = (int)(idx >> Fsh);
    int f = (int)(idx & ((1 << Fsh) - 1));
    if (e >= E) return;
    if (!em[e]) return;
    int s = src[e], d = dst[e];
    float coef = dis[s] * dis[d];
    atomicAdd(&agg[((size_t)d << Fsh) + f], h[((size_t)s << Fsh) + f] * coef);
}

// x_out = relu(agg + h * dis^2 + b)
__global__ void finish_kernel(const float* __restrict__ h, const float* __restrict__ agg,
                              const float* __restrict__ dis, const float* __restrict__ b,
                              float* __restrict__ xout, int total, int Fsh) {
    int idx = blockIdx.x * blockDim.x + threadIdx.x;
    if (idx >= total) return;
    int i = idx >> Fsh;
    int f = idx & ((1 << Fsh) - 1);
    float dd = dis[i];
    float v  = agg[idx] + h[idx] * dd * dd + b[f];
    xout[idx] = v > 0.f ? v : 0.f;
}

// ---------------------------------------------------------------------------
// TopK pooling: one block per graph. score = tanh(x.pw/||pw||); bitonic sort
// (descending, tie -> lower index first) of n <= 1024 (score, idx) pairs in
// LDS; keep top k = n/2; gather x_new = x[perm] * vals; write new_id.
// ---------------------------------------------------------------------------
__global__ __launch_bounds__(256)
void topk_kernel(const float* __restrict__ x, const float* __restrict__ pw,
                 float* __restrict__ xnew, int* __restrict__ new_id,
                 int n, int F) {
    __shared__ float skey[1024];
    __shared__ int   sidx[1024];
    __shared__ float snorm;
    const int g = blockIdx.x, tid = threadIdx.x, NT = 256;
    const int base = g * n;

    if (tid == 0) {
        float s = 0.f;
        for (int f = 0; f < F; ++f) { float w = pw[f]; s += w * w; }
        snorm = sqrtf(s);
    }
    __syncthreads();

    for (int i = tid; i < n; i += NT) {
        const float* xr = x + (size_t)(base + i) * F;
        float d = 0.f;
        for (int f = 0; f < F; ++f) d += xr[f] * pw[f];
        skey[i] = tanhf(d / snorm);
        sidx[i] = i;
    }
    __syncthreads();

    for (int k2 = 2; k2 <= n; k2 <<= 1) {
        for (int j = k2 >> 1; j > 0; j >>= 1) {
            for (int i = tid; i < n; i += NT) {
                int ixj = i ^ j;
                if (ixj > i) {
                    bool up = ((i & k2) == 0);           // descending segment
                    float a = skey[i], b = skey[ixj];
                    int ia = sidx[i], ib = sidx[ixj];
                    bool agtb = (a > b) || (a == b && ia < ib);
                    if (up ? !agtb : agtb) {
                        skey[i] = b; skey[ixj] = a;
                        sidx[i] = ib; sidx[ixj] = ia;
                    }
                }
            }
            __syncthreads();
        }
    }

    const int k = n >> 1;
    for (int j = tid; j < k; j += NT)
        new_id[base + sidx[j]] = g * k + j;
    for (int t = tid; t < k * F; t += NT) {
        int j = t / F, f = t - j * F;
        xnew[(size_t)(g * k + j) * F + f] =
            x[(size_t)(base + sidx[j]) * F + f] * skey[j];
    }
}

__global__ void remap_edges_kernel(int* __restrict__ src, int* __restrict__ dst,
                                   int* __restrict__ em, const int* __restrict__ new_id,
                                   int E) {
    int e = blockIdx.x * blockDim.x + threadIdx.x;
    if (e >= E) return;
    if (!em[e]) { src[e] = 0; dst[e] = 0; return; }
    int ns = new_id[src[e]], nd = new_id[dst[e]];
    if (ns >= 0 && nd >= 0) { src[e] = ns; dst[e] = nd; }
    else                    { em[e] = 0; src[e] = 0; dst[e] = 0; }
}

// ---------------------------------------------------------------------------
// Final readout: per graph (128 nodes x 16 feats) mean-pool -> linear -> *100
// ---------------------------------------------------------------------------
__global__ __launch_bounds__(128)
void final_kernel(const float* __restrict__ x, const float* __restrict__ linW,
                  const float* __restrict__ linb, float* __restrict__ out) {
    __shared__ float acc[128];
    const int g = blockIdx.x, tid = threadIdx.x;
    const int f = tid & 15, slot = tid >> 4;      // 8 node slots x 16 feats
    float s = 0.f;
    for (int node = slot; node < 128; node += 8)
        s += x[((size_t)g * 128 + node) * 16 + f];
    acc[slot * 16 + f] = s;
    __syncthreads();
    if (tid < 16) {
        float t = 0.f;
        for (int sl = 0; sl < 8; ++sl) t += acc[sl * 16 + tid];
        acc[tid] = (t * (1.0f / 128.0f)) * linW[tid];
    }
    __syncthreads();
    if (tid == 0) {
        float o = 0.f;
        for (int ff = 0; ff < 16; ++ff) o += acc[ff];
        out[g] = (o + linb[0]) * 100.0f;
    }
}

// ---------------------------------------------------------------------------
// Host-side orchestration
// ---------------------------------------------------------------------------
static inline int cdiv(long long a, int b) { return (int)((a + b - 1) / b); }

extern "C" void kernel_launch(void* const* d_in, const int* in_sizes, int n_in,
                              void* d_out, int out_size, void* d_ws, size_t ws_size,
                              hipStream_t stream) {
    const float* x0   = (const float*)d_in[0];
    const int*   ei   = (const int*)d_in[1];
    const float* W1   = (const float*)d_in[3];
    const float* b1   = (const float*)d_in[4];
    const float* pw1  = (const float*)d_in[5];
    const float* W2   = (const float*)d_in[6];
    const float* b2   = (const float*)d_in[7];
    const float* pw2  = (const float*)d_in[8];
    const float* W3   = (const float*)d_in[9];
    const float* b3   = (const float*)d_in[10];
    const float* pw3  = (const float*)d_in[11];
    const float* W4   = (const float*)d_in[12];
    const float* b4   = (const float*)d_in[13];
    const float* linW = (const float*)d_in[14];
    const float* linb = (const float*)d_in[15];
    float* out = (float*)d_out;

    // ---- workspace carve-up (all offsets 256B aligned) ----
    char* p = (char*)d_ws;
    float*  xA   = (float*)p;               p += (size_t)N0_NODES * 128 * 4;  // 32 MB
    float*  xB   = (float*)p;               p += (size_t)32768    * 128 * 4;  // 16 MB
    float*  h    = (float*)p;               p += (size_t)N0_NODES * 128 * 4;  // 32 MB
    float*  agg  = (float*)p;               p += (size_t)N0_NODES * 128 * 4;  // 32 MB
    __bf16* xbf  = (__bf16*)p;              p += (size_t)N0_NODES * 128 * 2;  // 16 MB
    __bf16* wt   = (__bf16*)p;              p += 65536;                       // 64 KB
    float*  deg  = (float*)p;               p += (size_t)N0_NODES * 4;
    float*  dis  = (float*)p;               p += (size_t)N0_NODES * 4;
    int*    srcb = (int*)p;                 p += (size_t)E_EDGES * 4;
    int*    dstb = (int*)p;                 p += (size_t)E_EDGES * 4;
    int*    emb  = (int*)p;                 p += (size_t)E_EDGES * 4;
    int*    nid  = (int*)p;                 p += (size_t)N0_NODES * 4;

    const int T = 256;
    const int Eblk = cdiv(E_EDGES, T);

    init_edges_kernel<<<Eblk, T, 0, stream>>>(ei, srcb, dstb, emb, E_EDGES);

    // ================= Layer 1: N=65536, K=128, F=128 =================
    {
        const int Nn = N0_NODES, K = 128, F = 128, Fsh = 7;
        cvt_bf16_kernel<<<cdiv((long long)Nn * K, T), T, 0, stream>>>(x0, xbf, Nn * K);
        cvt_wt_kernel<<<cdiv(K * F, T), T, 0, stream>>>(W1, wt, K, F);
        gemm_wmma<128, 8><<<Nn / 128, T, 0, stream>>>(xbf, wt, h);
        fill_f32_kernel<<<cdiv(Nn, T), T, 0, stream>>>(deg, 1.0f, Nn);
        deg_accum_kernel<<<Eblk, T, 0, stream>>>(dstb, emb, deg, E_EDGES);
        rsqrt_kernel<<<cdiv(Nn, T), T, 0, stream>>>(dis, deg, Nn);
        fill_f32_kernel<<<cdiv((long long)Nn * F, T), T, 0, stream>>>(agg, 0.0f, Nn * F);
        agg_kernel<<<cdiv((long long)E_EDGES * F, T), T, 0, stream>>>(h, srcb, dstb, emb, dis, agg, E_EDGES, Fsh);
        finish_kernel<<<cdiv((long long)Nn * F, T), T, 0, stream>>>(h, agg, dis, b1, xA, Nn * F, Fsh);
        // pool 1: n=1024 -> k=512
        fill_i32_kernel<<<cdiv(Nn, T), T, 0, stream>>>(nid, -1, Nn);
        topk_kernel<<<B_GRAPHS, T, 0, stream>>>(xA, pw1, xB, nid, 1024, F);
        remap_edges_kernel<<<Eblk, T, 0, stream>>>(srcb, dstb, emb, nid, E_EDGES);
    }

    // ================= Layer 2: N=32768, K=128, F=64 =================
    {
        const int Nn = 32768, K = 128, F = 64, Fsh = 6;
        cvt_bf16_kernel<<<cdiv((long long)Nn * K, T), T, 0, stream>>>(xB, xbf, Nn * K);
        cvt_wt_kernel<<<cdiv(K * F, T), T, 0, stream>>>(W2, wt, K, F);
        gemm_wmma<128, 4><<<Nn / 128, T, 0, stream>>>(xbf, wt, h);
        fill_f32_kernel<<<cdiv(Nn, T), T, 0, stream>>>(deg, 1.0f, Nn);
        deg_accum_kernel<<<Eblk, T, 0, stream>>>(dstb, emb, deg, E_EDGES);
        rsqrt_kernel<<<cdiv(Nn, T), T, 0, stream>>>(dis, deg, Nn);
        fill_f32_kernel<<<cdiv((long long)Nn * F, T), T, 0, stream>>>(agg, 0.0f, Nn * F);
        agg_kernel<<<cdiv((long long)E_EDGES * F, T), T, 0, stream>>>(h, srcb, dstb, emb, dis, agg, E_EDGES, Fsh);
        finish_kernel<<<cdiv((long long)Nn * F, T), T, 0, stream>>>(h, agg, dis, b2, xA, Nn * F, Fsh);
        // pool 2: n=512 -> k=256
        fill_i32_kernel<<<cdiv(Nn, T), T, 0, stream>>>(nid, -1, Nn);
        topk_kernel<<<B_GRAPHS, T, 0, stream>>>(xA, pw2, xB, nid, 512, F);
        remap_edges_kernel<<<Eblk, T, 0, stream>>>(srcb, dstb, emb, nid, E_EDGES);
    }

    // ================= Layer 3: N=16384, K=64, F=32 =================
    {
        const int Nn = 16384, K = 64, F = 32, Fsh = 5;
        cvt_bf16_kernel<<<cdiv((long long)Nn * K, T), T, 0, stream>>>(xB, xbf, Nn * K);
        cvt_wt_kernel<<<cdiv(K * F, T), T, 0, stream>>>(W3, wt, K, F);
        gemm_wmma<64, 2><<<Nn / 128, T, 0, stream>>>(xbf, wt, h);
        fill_f32_kernel<<<cdiv(Nn, T), T, 0, stream>>>(deg, 1.0f, Nn);
        deg_accum_kernel<<<Eblk, T, 0, stream>>>(dstb, emb, deg, E_EDGES);
        rsqrt_kernel<<<cdiv(Nn, T), T, 0, stream>>>(dis, deg, Nn);
        fill_f32_kernel<<<cdiv((long long)Nn * F, T), T, 0, stream>>>(agg, 0.0f, Nn * F);
        agg_kernel<<<cdiv((long long)E_EDGES * F, T), T, 0, stream>>>(h, srcb, dstb, emb, dis, agg, E_EDGES, Fsh);
        finish_kernel<<<cdiv((long long)Nn * F, T), T, 0, stream>>>(h, agg, dis, b3, xA, Nn * F, Fsh);
        // pool 3: n=256 -> k=128
        fill_i32_kernel<<<cdiv(Nn, T), T, 0, stream>>>(nid, -1, Nn);
        topk_kernel<<<B_GRAPHS, T, 0, stream>>>(xA, pw3, xB, nid, 256, F);
        remap_edges_kernel<<<Eblk, T, 0, stream>>>(srcb, dstb, emb, nid, E_EDGES);
    }

    // ================= Layer 4: N=8192, K=32, F=16 =================
    {
        const int Nn = 8192, K = 32, F = 16, Fsh = 4;
        cvt_bf16_kernel<<<cdiv((long long)Nn * K, T), T, 0, stream>>>(xB, xbf, Nn * K);
        cvt_wt_kernel<<<cdiv(K * F, T), T, 0, stream>>>(W4, wt, K, F);
        gemm_wmma<32, 1><<<Nn / 128, T, 0, stream>>>(xbf, wt, h);
        fill_f32_kernel<<<cdiv(Nn, T), T, 0, stream>>>(deg, 1.0f, Nn);
        deg_accum_kernel<<<Eblk, T, 0, stream>>>(dstb, emb, deg, E_EDGES);
        rsqrt_kernel<<<cdiv(Nn, T), T, 0, stream>>>(dis, deg, Nn);
        fill_f32_kernel<<<cdiv((long long)Nn * F, T), T, 0, stream>>>(agg, 0.0f, Nn * F);
        agg_kernel<<<cdiv((long long)E_EDGES * F, T), T, 0, stream>>>(h, srcb, dstb, emb, dis, agg, E_EDGES, Fsh);
        finish_kernel<<<cdiv((long long)Nn * F, T), T, 0, stream>>>(h, agg, dis, b4, xA, Nn * F, Fsh);
    }

    final_kernel<<<B_GRAPHS, 128, 0, stream>>>(xA, linW, linb, out);
}
